// PoET_88149908783430
// MI455X (gfx1250) — compile-verified, matrix operand order checked by reading, not run on hardware
//
#include <hip/hip_runtime.h>
#include <math.h>

typedef __bf16 bf16;
typedef __attribute__((ext_vector_type(16))) __bf16 v16bf;
typedef __attribute__((ext_vector_type(8)))  float  v8f;
typedef unsigned v4u __attribute__((ext_vector_type(4)));
typedef int      v4i __attribute__((ext_vector_type(4)));
typedef int      v8i __attribute__((ext_vector_type(8)));

#if __has_builtin(__builtin_amdgcn_tensor_load_to_lds) && __has_builtin(__builtin_amdgcn_s_wait_tensorcnt)
#define USE_TDM 1
#endif

static constexpr int Bb = 8, Ls = 16, D = 1024, H = 16, Dh = 64, S = 2048, DF = 4096, NL = 2;
static constexpr int M = Bb * Ls;            // 128 token rows
static constexpr float LN_EPS = 1e-5f;

// ---------------------------------------------------------------------------
// helpers
// ---------------------------------------------------------------------------
__device__ __forceinline__ int frag_k0(int v, int kbs) {
    // ISA 7.12.2: 16-bit A/B fragment K index for VGPR v; kbs = (lane>>4)*8
    return (v < 4) ? (kbs + 2 * v) : (16 + kbs + 2 * (v - 4));
}
__device__ __forceinline__ void set_pair(v16bf& f, int e, unsigned u) {
    bf16 two[2];
    __builtin_memcpy(two, &u, 4);
    f[e] = two[0]; f[e + 1] = two[1];
}
__device__ __forceinline__ v8f zero8() {
    v8f z;
#pragma unroll
    for (int i = 0; i < 8; ++i) z[i] = 0.f;
    return z;
}
#define WMMA_BF16(A_, B_, C_) \
    __builtin_amdgcn_wmma_f32_16x16x32_bf16(false, (A_), false, (B_), (short)0, (C_), false, false)

// ---------------------------------------------------------------------------
// LayerNorm: x (M,D) f32 -> out (M,D) bf16      grid=M, block=256
// ---------------------------------------------------------------------------
__global__ __launch_bounds__(256) void ln_kernel(const float* __restrict__ x,
                                                 const float* __restrict__ g,
                                                 const float* __restrict__ b,
                                                 bf16* __restrict__ out) {
    int row = blockIdx.x, tid = threadIdx.x;
    const float* xr = x + (size_t)row * D;
    float vals[4], s0 = 0.f, s1 = 0.f;
#pragma unroll
    for (int i = 0; i < 4; ++i) {
        float v = xr[tid + i * 256];
        vals[i] = v; s0 += v; s1 += v * v;
    }
#pragma unroll
    for (int m = 16; m >= 1; m >>= 1) {
        s0 += __shfl_down(s0, m, 32);
        s1 += __shfl_down(s1, m, 32);
    }
    __shared__ float r0[8], r1[8];
    if ((tid & 31) == 0) { r0[tid >> 5] = s0; r1[tid >> 5] = s1; }
    __syncthreads();
    float tot = 0.f, tot2 = 0.f;
#pragma unroll
    for (int j = 0; j < 8; ++j) { tot += r0[j]; tot2 += r1[j]; }
    float mean = tot * (1.0f / D);
    float var  = tot2 * (1.0f / D) - mean * mean;
    float rstd = rsqrtf(var + LN_EPS);
#pragma unroll
    for (int i = 0; i < 4; ++i) {
        int c = tid + i * 256;
        out[(size_t)row * D + c] = (bf16)((vals[i] - mean) * rstd * g[c] + b[c]);
    }
}

// ---------------------------------------------------------------------------
// Generic WMMA GEMM: C(M,N) = A_bf16(M,K) @ W_f32(K,N)  [+bias][gelu][+res]
// grid = (N/64, M/64), block = 128 (4 waves). Wave w owns rows w*16..w*16+15.
// A tile is DMA'd by the Tensor Data Mover (with HW row padding -> [64][72]);
// W tile is converted f32->bf16 in-flight by the VALU.
// ---------------------------------------------------------------------------
__global__ __launch_bounds__(128) void gemm_kernel(const bf16*  __restrict__ A,
                                                   const float* __restrict__ W,
                                                   const float* __restrict__ bias,
                                                   const float* __restrict__ res,
                                                   float* __restrict__ outF,
                                                   bf16*  __restrict__ outB,
                                                   int N, int K, int gelu) {
    __shared__ bf16 As[64][72];   // [m][k] (72 = 64 + 4-DWORD pad per row)
    __shared__ bf16 Ws[64][72];   // transposed: [n][k]
    const int tid = threadIdx.x;
    const int wv = tid >> 5, lane = tid & 31;
    const int nl = lane & 15, hi = lane >> 4, kbs = hi * 8;
    const int nb = blockIdx.x * 64, mb0 = blockIdx.y * 64;
    const int lr = tid >> 1, lh = tid & 1;

#ifdef USE_TDM
    const unsigned ldsAs = (unsigned)(unsigned long long)(void*)&As[0][0];
#endif

    v8f acc[4];
#pragma unroll
    for (int n = 0; n < 4; ++n) acc[n] = zero8();

    for (int kb = 0; kb < K; kb += 64) {
#ifdef USE_TDM
        if (tid < 32) {
            // TDM descriptor (ISA 8.3/8.4): 2-D tile 64x64, data_size=2B,
            // pad_interval = 32 DWORDs (code 4), pad_amount = 4 DWORDs (code 3)
            unsigned long long ga =
                (unsigned long long)(uintptr_t)(A + (size_t)mb0 * K + kb);
            v4u g0;
            g0[0] = 1u;                                   // count=1, user mode
            g0[1] = ldsAs;                                // lds_addr
            g0[2] = (unsigned)ga;                         // global_addr[31:0]
            g0[3] = (unsigned)((ga >> 32) & 0x01FFFFFFu)  // global_addr[56:32]
                  | (2u << 30);                           // type = 2 (image)
            v8i g1;
            g1[0] = (int)((1u << 16)      // data_size = 1 -> 2 bytes
                        | (1u << 20)      // pad_enable
                        | (4u << 22)      // pad_interval code 4 -> 32 DWORDs
                        | (3u << 25));    // pad_amount  code 3 -> 4 DWORDs
            g1[1] = (int)(64u << 16);     // tensor_dim0 = 64 (low 16 in [31:16])
            g1[2] = (int)(64u << 16);     // tensor_dim1 = 64
            g1[3] = (int)(64u << 16);     // tile_dim0 = 64
            g1[4] = 64;                   // tile_dim1 = 64 (tile_dim2 = 0)
            g1[5] = K;                    // tensor_dim0_stride[31:0]
            g1[6] = 0;                    // stride0 hi / stride1 lo
            g1[7] = 0;
            v4i gz4; gz4[0] = 0; gz4[1] = 0; gz4[2] = 0; gz4[3] = 0;
            v8i gz8;
#pragma unroll
            for (int z = 0; z < 8; ++z) gz8[z] = 0;
            // amdgpu-toolchain (clang-23) 6-arg form
            __builtin_amdgcn_tensor_load_to_lds(g0, g1, gz4, gz4, gz8, 0);
        }
#else
        {   // fallback: cooperative vector load, 8 x uint2 per thread
            const uint2* gA = (const uint2*)(A + (size_t)(mb0 + lr) * K + kb + lh * 32);
            uint2* sA = (uint2*)&As[lr][lh * 32];
#pragma unroll
            for (int j = 0; j < 8; ++j) sA[j] = gA[j];
        }
#endif
        {   // W tile: 64x64 f32 -> bf16, stored transposed [n][k]
            const float4* gW = (const float4*)(W + (size_t)(kb + lr) * N + nb + lh * 32);
#pragma unroll
            for (int j = 0; j < 8; ++j) {
                float4 f = gW[j];
                int n0 = lh * 32 + j * 4;
                Ws[n0 + 0][lr] = (bf16)f.x;
                Ws[n0 + 1][lr] = (bf16)f.y;
                Ws[n0 + 2][lr] = (bf16)f.z;
                Ws[n0 + 3][lr] = (bf16)f.w;
            }
        }
        if (kb + 64 < K)
            __builtin_prefetch(W + (size_t)(kb + 64 + lr) * N + nb, 0, 1);
#ifdef USE_TDM
        if (tid < 32) __builtin_amdgcn_s_wait_tensorcnt(0);
#endif
        __syncthreads();

        const int mloc = wv * 16 + nl;
#pragma unroll
        for (int kh = 0; kh < 2; ++kh) {
            v16bf af;
#pragma unroll
            for (int v = 0; v < 8; ++v) {
                int k0 = frag_k0(v, kbs) + kh * 32;
                set_pair(af, 2 * v, *(const unsigned*)&As[mloc][k0]);
            }
#pragma unroll
            for (int n = 0; n < 4; ++n) {
                v16bf bfrag;
                int nloc = n * 16 + nl;
#pragma unroll
                for (int v = 0; v < 8; ++v) {
                    int k0 = frag_k0(v, kbs) + kh * 32;
                    set_pair(bfrag, 2 * v, *(const unsigned*)&Ws[nloc][k0]);
                }
                acc[n] = WMMA_BF16(af, bfrag, acc[n]);
            }
        }
        __syncthreads();
    }
    // epilogue: C layout — VGPR r, lane: row = r + hi*8, col = n*16 + nl
#pragma unroll
    for (int n = 0; n < 4; ++n) {
        int col = nb + n * 16 + nl;
#pragma unroll
        for (int r = 0; r < 8; ++r) {
            int row = mb0 + wv * 16 + r + hi * 8;
            float v = acc[n][r];
            if (bias) v += bias[col];
            if (gelu) {
                float x3 = v * v * v;
                v = 0.5f * v * (1.0f + tanhf(0.7978845608028654f * (v + 0.044715f * x3)));
            }
            if (res)  v += res[(size_t)row * N + col];
            if (outF) outF[(size_t)row * N + col] = v;
            if (outB) outB[(size_t)row * N + col] = (bf16)v;
        }
    }
}

// ---------------------------------------------------------------------------
// RoPE on q,k (f32, (B,L,H,Dh)) -> bf16 in (B,H,L,Dh) layout. pos = S + l.
// ---------------------------------------------------------------------------
__global__ __launch_bounds__(256) void rope_kernel(const float* __restrict__ qf,
                                                   const float* __restrict__ kf,
                                                   bf16* __restrict__ qo,
                                                   bf16* __restrict__ ko) {
    int idx = blockIdx.x * 256 + threadIdx.x;       // B*L*H*32
    if (idx >= Bb * Ls * H * 32) return;
    int j = idx & 31, h = (idx >> 5) & 15, l = (idx >> 9) & 15, b = idx >> 13;
    float inv = __expf(-(float)j * (logf(10000.0f) / 32.0f));
    float f = (float)(S + l) * inv;
    float c = cosf(f), s = sinf(f);
    size_t ib = (((size_t)b * Ls + l) * H + h) * Dh;
    size_t ob = (((size_t)b * H + h) * Ls + l) * Dh;
    float q1 = qf[ib + j], q2 = qf[ib + j + 32];
    float k1 = kf[ib + j], k2 = kf[ib + j + 32];
    qo[ob + j]      = (bf16)(q1 * c - q2 * s);
    qo[ob + j + 32] = (bf16)(q2 * c + q1 * s);
    ko[ob + j]      = (bf16)(k1 * c - k2 * s);
    ko[ob + j + 32] = (bf16)(k2 * c + k1 * s);
}

// memory K: (S,H,Dh) f32 -> (H,S,Dh) bf16
__global__ __launch_bounds__(256) void conv_kmem(const float* __restrict__ km, bf16* __restrict__ out) {
    int idx = blockIdx.x * 256 + threadIdx.x;
    if (idx >= S * H * Dh) return;
    int d = idx & 63, h = (idx >> 6) & 15, t = idx >> 10;
    out[((size_t)h * S + t) * Dh + d] = (bf16)km[idx];
}
// memory V: (S,H,Dh) f32 -> (H,Dh,S) bf16 (transposed for B-fragments)
__global__ __launch_bounds__(256) void conv_vmemT(const float* __restrict__ vm, bf16* __restrict__ out) {
    int idx = blockIdx.x * 256 + threadIdx.x;
    if (idx >= S * H * Dh) return;
    int d = idx & 63, h = (idx >> 6) & 15, t = idx >> 10;
    out[((size_t)h * Dh + d) * S + t] = (bf16)vm[idx];
}
// new V: (B,L,H,Dh) f32 -> (B,H,Dh,L) bf16
__global__ __launch_bounds__(256) void conv_vnewT(const float* __restrict__ vf, bf16* __restrict__ out) {
    int idx = blockIdx.x * 256 + threadIdx.x;
    if (idx >= Bb * Ls * H * Dh) return;
    int d = idx & 63, h = (idx >> 6) & 15, l = (idx >> 10) & 15, b = idx >> 14;
    out[(((size_t)b * H + h) * Dh + d) * Ls + l] = (bf16)vf[idx];
}

// ---------------------------------------------------------------------------
// Flash attention, one block per (b,h). 4 waves; wave w streams 32 keys/chunk.
// T = S + L = 2064 = 16 chunks of 128 + one 16-wide tail (the new keys).
// ---------------------------------------------------------------------------
__global__ __launch_bounds__(128) void attn_kernel(const bf16* __restrict__ q_bf,   // (B,H,L,Dh)
                                                   const bf16* __restrict__ kmem,   // (H,S,Dh)
                                                   const bf16* __restrict__ vmemT,  // (H,Dh,S)
                                                   const bf16* __restrict__ knew,   // (B,H,L,Dh)
                                                   const bf16* __restrict__ vnewT,  // (B,H,Dh,L)
                                                   bf16* __restrict__ out) {        // (M,D)
    const int bh = blockIdx.x, h = bh & 15, b = bh >> 4;
    const int tid = threadIdx.x, w = tid >> 5, lane = tid & 31;
    const int nl = lane & 15, hi = lane >> 4, kbs = hi * 8;
    const float NEG = -__builtin_inff();

    __shared__ bf16  pl[4][16][40];     // per-wave staged P tile (16 x 32)
    __shared__ float Osh[4][16][64];
    __shared__ float msh[4][16], ssh[4][16];

    // q A-fragments (rows = L, K = Dh in two halves)
    const bf16* qrow = q_bf + (((size_t)b * H + h) * Ls + nl) * Dh;
    v16bf aq0, aq1;
#pragma unroll
    for (int v = 0; v < 8; ++v) {
        int k0 = frag_k0(v, kbs);
        set_pair(aq0, 2 * v, *(const unsigned*)&qrow[k0]);
        set_pair(aq1, 2 * v, *(const unsigned*)&qrow[32 + k0]);
    }

    float mrow[8], srow[8];
    v8f Oacc[4];
#pragma unroll
    for (int r = 0; r < 8; ++r) { mrow[r] = NEG; srow[r] = 0.f; }
#pragma unroll
    for (int n = 0; n < 4; ++n) Oacc[n] = zero8();

    for (int ic = 0; ic <= 16; ++ic) {
        const bool tail = (ic == 16);
        const int t0 = ic * 128 + w * 32;
        const int valid = tail ? (w == 0 ? 16 : 0) : 32;

        v8f st0, st1;
        if (valid == 32) {
            const bf16* Kb = kmem + ((size_t)h * S + t0) * Dh;
            v16bf bk;
            st0 = zero8();
#pragma unroll
            for (int v = 0; v < 8; ++v) set_pair(bk, 2 * v, *(const unsigned*)&Kb[(size_t)nl * Dh + frag_k0(v, kbs)]);
            st0 = WMMA_BF16(aq0, bk, st0);
#pragma unroll
            for (int v = 0; v < 8; ++v) set_pair(bk, 2 * v, *(const unsigned*)&Kb[(size_t)nl * Dh + 32 + frag_k0(v, kbs)]);
            st0 = WMMA_BF16(aq1, bk, st0);
            const bf16* Kb2 = Kb + (size_t)16 * Dh;
            st1 = zero8();
#pragma unroll
            for (int v = 0; v < 8; ++v) set_pair(bk, 2 * v, *(const unsigned*)&Kb2[(size_t)nl * Dh + frag_k0(v, kbs)]);
            st1 = WMMA_BF16(aq0, bk, st1);
#pragma unroll
            for (int v = 0; v < 8; ++v) set_pair(bk, 2 * v, *(const unsigned*)&Kb2[(size_t)nl * Dh + 32 + frag_k0(v, kbs)]);
            st1 = WMMA_BF16(aq1, bk, st1);
        } else if (valid == 16) {
            const bf16* Kb = knew + (((size_t)b * H + h) * Ls) * Dh;
            v16bf bk;
            st0 = zero8();
#pragma unroll
            for (int v = 0; v < 8; ++v) set_pair(bk, 2 * v, *(const unsigned*)&Kb[(size_t)nl * Dh + frag_k0(v, kbs)]);
            st0 = WMMA_BF16(aq0, bk, st0);
#pragma unroll
            for (int v = 0; v < 8; ++v) set_pair(bk, 2 * v, *(const unsigned*)&Kb[(size_t)nl * Dh + 32 + frag_k0(v, kbs)]);
            st0 = WMMA_BF16(aq1, bk, st0);
#pragma unroll
            for (int r = 0; r < 8; ++r) st1[r] = NEG;
        } else {
#pragma unroll
            for (int r = 0; r < 8; ++r) { st0[r] = NEG; st1[r] = NEG; }
        }
        // scale (−inf stays −inf)
#pragma unroll
        for (int r = 0; r < 8; ++r) { st0[r] *= 0.125f; st1[r] *= 0.125f; }

        // per-row max across the 32 columns of this wave's tile
        float tm[8];
#pragma unroll
        for (int r = 0; r < 8; ++r) tm[r] = fmaxf(st0[r], st1[r]);
#pragma unroll
        for (int mk = 1; mk <= 8; mk <<= 1)
#pragma unroll
            for (int r = 0; r < 8; ++r) tm[r] = fmaxf(tm[r], __shfl_xor(tm[r], mk, 32));

        float al[8];
#pragma unroll
        for (int r = 0; r < 8; ++r) {
            float mn = fmaxf(mrow[r], tm[r]);
            al[r] = __expf(mrow[r] - mn);
            mrow[r] = mn;
        }
#pragma unroll
        for (int r = 0; r < 8; ++r) srow[r] *= al[r];
#pragma unroll
        for (int n = 0; n < 4; ++n)
#pragma unroll
            for (int r = 0; r < 8; ++r) Oacc[n][r] *= al[r];

        float p0[8], p1[8], ps[8];
#pragma unroll
        for (int r = 0; r < 8; ++r) {
            p0[r] = __expf(st0[r] - mrow[r]);
            p1[r] = __expf(st1[r] - mrow[r]);
            ps[r] = p0[r] + p1[r];
        }
#pragma unroll
        for (int mk = 1; mk <= 8; mk <<= 1)
#pragma unroll
            for (int r = 0; r < 8; ++r) ps[r] += __shfl_xor(ps[r], mk, 32);
#pragma unroll
        for (int r = 0; r < 8; ++r) srow[r] += ps[r];

        // stage P (16x32) bf16 in this wave's LDS region
#pragma unroll
        for (int r = 0; r < 8; ++r) {
            int row = r + hi * 8;
            pl[w][row][nl]      = (bf16)p0[r];
            pl[w][row][16 + nl] = (bf16)p1[r];
        }
        __builtin_amdgcn_wave_barrier();

        if (valid) {
            v16bf ap;
#pragma unroll
            for (int v = 0; v < 8; ++v)
                set_pair(ap, 2 * v, *(const unsigned*)&pl[w][nl][frag_k0(v, kbs)]);
#pragma unroll
            for (int n = 0; n < 4; ++n) {
                v16bf bv;
                int d = n * 16 + nl;
                if (!tail) {
                    const bf16* Vb = vmemT + ((size_t)h * Dh + d) * S + t0;
#pragma unroll
                    for (int v = 0; v < 8; ++v) set_pair(bv, 2 * v, *(const unsigned*)&Vb[frag_k0(v, kbs)]);
                } else {
                    const bf16* Vb = vnewT + (((size_t)b * H + h) * Dh + d) * Ls;
#pragma unroll
                    for (int v = 0; v < 8; ++v) {
                        int k0 = frag_k0(v, kbs);
                        unsigned u = (k0 < 16) ? *(const unsigned*)&Vb[k0] : 0u;
                        set_pair(bv, 2 * v, u);
                    }
                }
                Oacc[n] = WMMA_BF16(ap, bv, Oacc[n]);
            }
        }
    }

    // write per-wave partials and merge
#pragma unroll
    for (int n = 0; n < 4; ++n)
#pragma unroll
        for (int r = 0; r < 8; ++r)
            Osh[w][r + hi * 8][n * 16 + nl] = Oacc[n][r];
    if (nl == 0)
#pragma unroll
        for (int r = 0; r < 8; ++r) {
            msh[w][r + hi * 8] = mrow[r];
            ssh[w][r + hi * 8] = srow[r];
        }
    __syncthreads();

    for (int idx = tid; idx < 16 * 64; idx += 128) {
        int row = idx >> 6, col = idx & 63;
        float mf = NEG;
#pragma unroll
        for (int ww = 0; ww < 4; ++ww) mf = fmaxf(mf, msh[ww][row]);
        float sf = 0.f, of = 0.f;
#pragma unroll
        for (int ww = 0; ww < 4; ++ww) {
            float e = __expf(msh[ww][row] - mf);
            sf += ssh[ww][row] * e;
            of += Osh[ww][row][col] * e;
        }
        out[((size_t)(b * Ls + row)) * D + h * Dh + col] = (bf16)(of / sf);
    }
}

// ---------------------------------------------------------------------------
// host
// ---------------------------------------------------------------------------
extern "C" void kernel_launch(void* const* d_in, const int* in_sizes, int n_in,
                              void* d_out, int out_size, void* d_ws, size_t ws_size,
                              hipStream_t stream) {
    (void)in_sizes; (void)n_in; (void)out_size; (void)ws_size;
    const float* x_in       = (const float*)d_in[0];
    const float* self_k_mem = (const float*)d_in[1];
    const float* self_v_mem = (const float*)d_in[2];
    const float* cross_k_mem= (const float*)d_in[3];
    const float* cross_v_mem= (const float*)d_in[4];
    const float* self_Wq = (const float*)d_in[5];
    const float* self_Wk = (const float*)d_in[6];
    const float* self_Wv = (const float*)d_in[7];
    const float* self_Wo = (const float*)d_in[8];
    const float* cross_Wq= (const float*)d_in[9];
    const float* cross_Wk= (const float*)d_in[10];
    const float* cross_Wv= (const float*)d_in[11];
    const float* cross_Wo= (const float*)d_in[12];
    const float* ln1_g = (const float*)d_in[13];
    const float* ln1_b = (const float*)d_in[14];
    const float* ln2_g = (const float*)d_in[15];
    const float* ln2_b = (const float*)d_in[16];
    const float* ln3_g = (const float*)d_in[17];
    const float* ln3_b = (const float*)d_in[18];
    const float* W1 = (const float*)d_in[19];
    const float* b1 = (const float*)d_in[20];
    const float* W2 = (const float*)d_in[21];
    const float* b2 = (const float*)d_in[22];

    float* x = (float*)d_out;   // x lives in d_out; residual adds in-place

    // workspace layout
    unsigned char* w8 = (unsigned char*)d_ws;
    size_t off = 0;
    auto alloc = [&](size_t bytes) -> void* {
        void* p = w8 + off;
        off += (bytes + 255) & ~(size_t)255;
        return p;
    };
    bf16*  h_bf   = (bf16*)alloc((size_t)M * D * 2);
    float* qf     = (float*)alloc((size_t)M * D * 4);
    float* kf     = (float*)alloc((size_t)M * D * 4);
    float* vf     = (float*)alloc((size_t)M * D * 4);
    bf16*  q_bf   = (bf16*)alloc((size_t)M * D * 2);
    bf16*  knew   = (bf16*)alloc((size_t)M * D * 2);
    bf16*  vnewT  = (bf16*)alloc((size_t)M * D * 2);
    bf16*  kmemBf = (bf16*)alloc((size_t)S * H * Dh * 2);
    bf16*  vmemTb = (bf16*)alloc((size_t)S * H * Dh * 2);
    bf16*  attnBf = (bf16*)alloc((size_t)M * D * 2);
    bf16*  ffBf   = (bf16*)alloc((size_t)M * DF * 2);

    (void)hipMemcpyAsync(x, x_in, (size_t)M * D * sizeof(float), hipMemcpyDeviceToDevice, stream);

    const int memN = S * H * Dh;                    // 2M elements
    auto run_block = [&](const float* Wq, const float* Wk, const float* Wv, const float* Wo,
                         const float* g, const float* bb,
                         const float* km, const float* vm) {
        ln_kernel<<<M, 256, 0, stream>>>(x, g, bb, h_bf);
        gemm_kernel<<<dim3(16, 2), 128, 0, stream>>>(h_bf, Wq, nullptr, nullptr, qf, nullptr, D, D, 0);
        gemm_kernel<<<dim3(16, 2), 128, 0, stream>>>(h_bf, Wk, nullptr, nullptr, kf, nullptr, D, D, 0);
        gemm_kernel<<<dim3(16, 2), 128, 0, stream>>>(h_bf, Wv, nullptr, nullptr, vf, nullptr, D, D, 0);
        rope_kernel<<<(Bb * Ls * H * 32) / 256, 256, 0, stream>>>(qf, kf, q_bf, knew);
        conv_vnewT<<<(M * D) / 256, 256, 0, stream>>>(vf, vnewT);
        conv_kmem<<<memN / 256, 256, 0, stream>>>(km, kmemBf);
        conv_vmemT<<<memN / 256, 256, 0, stream>>>(vm, vmemTb);
        attn_kernel<<<Bb * H, 128, 0, stream>>>(q_bf, kmemBf, vmemTb, knew, vnewT, attnBf);
        gemm_kernel<<<dim3(16, 2), 128, 0, stream>>>(attnBf, Wo, nullptr, x, x, nullptr, D, D, 0);
    };

    for (int l = 0; l < NL; ++l) {
        size_t wOff = (size_t)l * D * D;
        size_t mOff = (size_t)l * S * H * Dh;
        run_block(self_Wq + wOff, self_Wk + wOff, self_Wv + wOff, self_Wo + wOff,
                  ln1_g + (size_t)l * D, ln1_b + (size_t)l * D,
                  self_k_mem + mOff, self_v_mem + mOff);
        run_block(cross_Wq + wOff, cross_Wk + wOff, cross_Wv + wOff, cross_Wo + wOff,
                  ln2_g + (size_t)l * D, ln2_b + (size_t)l * D,
                  cross_k_mem + mOff, cross_v_mem + mOff);
        ln_kernel<<<M, 256, 0, stream>>>(x, ln3_g + (size_t)l * D, ln3_b + (size_t)l * D, h_bf);
        gemm_kernel<<<dim3(64, 2), 128, 0, stream>>>(h_bf, W1 + (size_t)l * D * DF,
                                                     b1 + (size_t)l * DF, nullptr,
                                                     nullptr, ffBf, DF, D, 1);
        gemm_kernel<<<dim3(16, 2), 128, 0, stream>>>(ffBf, W2 + (size_t)l * DF * D,
                                                     b2 + (size_t)l * D, x,
                                                     x, nullptr, D, DF, 0);
    }
}
